// Hidformer_37288906064646
// MI455X (gfx1250) — matrix operand order repaired and checked by reading, not compiled, not run on hardware
//
#include <hip/hip_runtime.h>
#include <hip/hip_bf16.h>
#include <math.h>

// ---------------------------------------------------------------------------
// Problem constants (from reference)
// ---------------------------------------------------------------------------
#define B_SZ   32
#define T_SZ   4096
#define C_SZ   32
#define L_SZ   16
#define S_SZ   8
#define NSEG   511            // (T-L)/S + 1
#define NSEGP  512            // padded attention K/ld
#define D_DIM  512            // C*L
#define H_DIM  512
#define NF     256            // rfft bins of length-511 axis
#define U3     1536           // 3*H
#define FF     2048           // 4*D
#define OUTD   32
#define EPS_F  1e-5f

// ---------------------------------------------------------------------------
// WMMA types (CDNA5 / gfx1250, wave32)
// ---------------------------------------------------------------------------
typedef __attribute__((ext_vector_type(16))) __bf16 bf16x16;
typedef __attribute__((ext_vector_type(8)))  float  f32x8;
typedef __attribute__((ext_vector_type(4)))  float  f32x4;

union AFrag { bf16x16 v; unsigned int u[8]; };
union AccFr { f32x8 v; float f[8]; };

__device__ __forceinline__ unsigned int f2bf(float f) {
  unsigned int u = __float_as_uint(f);
  return (u + 0x7FFFu + ((u >> 16) & 1u)) >> 16;    // round-to-nearest-even
}
__device__ __forceinline__ unsigned int pkbf(float a, float b) {
  return f2bf(a) | (f2bf(b) << 16);
}

// ---------------------------------------------------------------------------
// Generic bf16-WMMA GEMM (requires Kd % 32 == 0, 16B-aligned rows):
//   C[m,n] = epi( scale * sum_k A[m,k] * B(k,n) ),  batched via z-grid.
//   BT=true : B stored as W[n][k]  (row-major NxK; computes A @ W^T)
//   BT=false: B stored as B[k][n]  (row-major KxN)
//   BW=true : B is pre-converted bf16 [n][k] (requires BT, Nd % 128 == 0);
//             staged with global_load_async_to_lds_b128 (ASYNCcnt path).
//   EPI: 0 = none, 1 = +bias[n], 2 = +bias[n] then exact GELU
// Block tile 128x128, K-step 32; 8 waves (4x2), 32(M)x64(N) per wave,
// i.e. 2x4 WMMA fragments -> 8 v_wmma per wave per K-step.
// ---------------------------------------------------------------------------
#define BM 128
#define BN 128
#define BK 32
#define LDSU 20   // uint (k-pair) stride per row: 16 pairs + 4 pad (80B, 16B-aligned)

template <bool BT, int EPI, bool BW>
__global__ __launch_bounds__(256)
void gemm_wmma_bf16(const float* __restrict__ A, const void* __restrict__ Bp,
                    const float* __restrict__ bias, float* __restrict__ Cc,
                    int M, int Nd, int Kd, int lda, int ldb, int ldc,
                    long long aBatch, long long bBatch, long long cBatch,
                    float scale) {
  __shared__ unsigned int As[BM][LDSU];
  __shared__ unsigned int Bs[BN][LDSU];

  const float*          Bm = (const float*)Bp;
  const unsigned short* Bw = (const unsigned short*)Bp;

  const int bz = blockIdx.z;
  A  += (size_t)bz * aBatch;
  Bm += (size_t)bz * bBatch;
  Bw += (size_t)bz * bBatch;
  Cc += (size_t)bz * cBatch;

  const int m0  = blockIdx.y * BM;
  const int n0  = blockIdx.x * BN;
  const int tid = threadIdx.x;
  const int wave = tid >> 5, lane = tid & 31;
  const int wm = wave >> 1, wn = wave & 1;     // 4 x 2 wave grid
  const int hl = lane >> 4, lr = lane & 15;    // half / lane-in-half

  // staging maps
  const int rT = tid >> 1;             // 0..127 (A row / B row for BT)
  const int cT = (tid & 1) * 16;       // k offset 0 / 16
  const int nB = tid & 127;            // BT=false: fixed n
  const int kB = (tid >> 7) * 16;      // BT=false: k group 0 / 16

  AccFr acc[2][4];
#pragma unroll
  for (int i = 0; i < 2; i++)
#pragma unroll
    for (int j = 0; j < 4; j++)
#pragma unroll
      for (int t = 0; t < 8; t++) acc[i][j].f[t] = 0.0f;

  for (int k0 = 0; k0 < Kd; k0 += BK) {
    // ---- stage B tile first (async path wants max overlap with A pack) ----
    if (BW) {
      // pure bf16 copy global->LDS, 2x b128 per thread, tracked by ASYNCcnt
      const int gn = n0 + rT;                       // Nd % 128 == 0: in range
      unsigned lds0 = (unsigned)(size_t)&Bs[rT][cT >> 1];
      unsigned long long ga =
          (unsigned long long)(size_t)(Bw + (size_t)gn * ldb + (k0 + cT));
      asm volatile("global_load_async_to_lds_b128 %0, %1, off"
                   :: "v"(lds0), "v"(ga) : "memory");
      asm volatile("global_load_async_to_lds_b128 %0, %1, off offset:16"
                   :: "v"(lds0 + 16u), "v"(ga) : "memory");
    } else if (BT) {
      const int gn = n0 + rT;
      const int gnc = (gn < Nd) ? gn : (Nd - 1);
      const f32x4* pb = (const f32x4*)(Bm + (size_t)gnc * ldb + (k0 + cT));
      f32x4 x0 = pb[0], x1 = pb[1], x2 = pb[2], x3 = pb[3];
      if (gn >= Nd) { x0 = 0.0f; x1 = 0.0f; x2 = 0.0f; x3 = 0.0f; }
      unsigned int* d = &Bs[rT][cT >> 1];
      d[0] = pkbf(x0[0], x0[1]); d[1] = pkbf(x0[2], x0[3]);
      d[2] = pkbf(x1[0], x1[1]); d[3] = pkbf(x1[2], x1[3]);
      d[4] = pkbf(x2[0], x2[1]); d[5] = pkbf(x2[2], x2[3]);
      d[6] = pkbf(x3[0], x3[1]); d[7] = pkbf(x3[2], x3[3]);
    } else {
      const int gn = n0 + nB;
      const int gnc = (gn < Nd) ? gn : (Nd - 1);
      const bool ok = gn < Nd;
      const float* pb = Bm + (size_t)(k0 + kB) * ldb + gnc;
      unsigned int* d = &Bs[nB][kB >> 1];
#pragma unroll
      for (int q = 0; q < 8; q++) {
        float v0 = pb[(size_t)(2 * q) * ldb];
        float v1 = pb[(size_t)(2 * q + 1) * ldb];
        if (!ok) { v0 = 0.0f; v1 = 0.0f; }
        d[q] = pkbf(v0, v1);
      }
    }
    // ---- stage A tile (128x32): 4x float4 per thread, pack bf16 pairs ----
    {
      const int gm = m0 + rT;
      const int gmc = (gm < M) ? gm : (M - 1);
      const f32x4* pa = (const f32x4*)(A + (size_t)gmc * lda + (k0 + cT));
      f32x4 x0 = pa[0], x1 = pa[1], x2 = pa[2], x3 = pa[3];
      if (gm >= M) { x0 = 0.0f; x1 = 0.0f; x2 = 0.0f; x3 = 0.0f; }
      unsigned int* d = &As[rT][cT >> 1];
      d[0] = pkbf(x0[0], x0[1]); d[1] = pkbf(x0[2], x0[3]);
      d[2] = pkbf(x1[0], x1[1]); d[3] = pkbf(x1[2], x1[3]);
      d[4] = pkbf(x2[0], x2[1]); d[5] = pkbf(x2[2], x2[3]);
      d[6] = pkbf(x3[0], x3[1]); d[7] = pkbf(x3[2], x3[3]);
    }
    if (BW) asm volatile("s_wait_asynccnt 0" ::: "memory");
    __syncthreads();

    // hint-prefetch next A K-slab (global_prefetch_b8)
    if (k0 + BK < Kd)
      __builtin_prefetch(A + (size_t)((m0 + rT < M) ? (m0 + rT) : (M - 1)) * lda + k0 + BK, 0, 1);

    // ---- fragments per ISA 16-bit layouts ----
    AFrag afr[2], bfr[4];
#pragma unroll
    for (int i = 0; i < 2; i++) {
      const int m = wm * 32 + i * 16 + lr;
#pragma unroll
      for (int v = 0; v < 8; v++)      // A 16x32: half0 K={0..7,16..23}, half1 +8
        afr[i].u[v] = As[m][(v < 4 ? v : 4 + v) + 4 * hl];
    }
#pragma unroll
    for (int j = 0; j < 4; j++) {
      const int n = wn * 64 + j * 16 + lr;
#pragma unroll
      for (int v = 0; v < 8; v++)      // B 32x16: half0 K={0..7,16..23}, half1 +8
        bfr[j].u[v] = Bs[n][(v < 4 ? v : 4 + v) + 4 * hl];
    }
#pragma unroll
    for (int i = 0; i < 2; i++)
#pragma unroll
      for (int j = 0; j < 4; j++)
        acc[i][j].v = __builtin_amdgcn_wmma_f32_16x16x32_bf16(
            false, afr[i].v, false, bfr[j].v, (short)0, acc[i][j].v, false, false);
    __syncthreads();
  }

  // ---- epilogue: C layout VGPR r -> row r+8*half, col lr ----
#pragma unroll
  for (int i = 0; i < 2; i++)
#pragma unroll
    for (int j = 0; j < 4; j++)
#pragma unroll
      for (int rr = 0; rr < 8; rr++) {
        const int row = m0 + wm * 32 + i * 16 + rr + 8 * hl;
        const int col = n0 + wn * 64 + j * 16 + lr;
        if (row < M && col < Nd) {
          float v = acc[i][j].f[rr] * scale;
          if (EPI >= 1) v += bias[col];
          if (EPI == 2) v = 0.5f * v * (1.0f + erff(v * 0.70710678118654752f));
          Cc[(size_t)row * ldc + col] = v;
        }
      }
}

// f32 -> bf16 weight pre-conversion
__global__ void cvt_bf16(const float* __restrict__ in, unsigned short* __restrict__ out,
                         int n) {
  int i = blockIdx.x * blockDim.x + threadIdx.x;
  if (i < n) out[i] = (unsigned short)f2bf(in[i]);
}

// ---------------------------------------------------------------------------
// RevIN statistics per (b, c): mean / (std(ddof=1)+eps) folded to scale/shift
// ---------------------------------------------------------------------------
__global__ __launch_bounds__(256)
void revin_stats(const float* __restrict__ x, const float* __restrict__ rw,
                 const float* __restrict__ rb, float* __restrict__ sc,
                 float* __restrict__ sh) {
  const int row = blockIdx.x;           // b*C + c
  const int b = row / C_SZ, c = row % C_SZ;
  float s = 0.f, s2 = 0.f;
  for (int t = threadIdx.x; t < T_SZ; t += blockDim.x) {
    float v = x[((size_t)b * T_SZ + t) * C_SZ + c];
    s += v; s2 += v * v;
  }
  __shared__ float r1[8], r2[8];
  for (int m = 16; m >= 1; m >>= 1) { s += __shfl_xor(s, m); s2 += __shfl_xor(s2, m); }
  const int wid = threadIdx.x >> 5;
  if ((threadIdx.x & 31) == 0) { r1[wid] = s; r2[wid] = s2; }
  __syncthreads();
  if (threadIdx.x == 0) {
    float S = 0.f, S2 = 0.f;
    for (int i = 0; i < 8; i++) { S += r1[i]; S2 += r2[i]; }
    float mean = S / (float)T_SZ;
    float var = (S2 - (float)T_SZ * mean * mean) / (float)(T_SZ - 1);
    var = fmaxf(var, 0.f);
    float inv = 1.0f / (sqrtf(var) + EPS_F);
    float a = rw[c] * inv;
    sc[row] = a;
    sh[row] = rb[c] - mean * a;
  }
}

// tokens[b,n,c*L+l] = x[b, n*S+l, c]*scale + shift
__global__ void build_tokens(const float* __restrict__ x, const float* __restrict__ sc,
                             const float* __restrict__ sh, float* __restrict__ tok) {
  size_t idx = (size_t)blockIdx.x * blockDim.x + threadIdx.x;
  const size_t total = (size_t)B_SZ * NSEG * D_DIM;
  if (idx >= total) return;
  int b = (int)(idx / ((size_t)NSEG * D_DIM));
  int rem = (int)(idx % ((size_t)NSEG * D_DIM));
  int n = rem / D_DIM, dd = rem % D_DIM;
  int c = dd / L_SZ, l = dd % L_SZ;
  int t = n * S_SZ + l;
  float v = x[((size_t)b * T_SZ + t) * C_SZ + c];
  int p = b * C_SZ + c;
  tok[idx] = v * sc[p] + sh[p];
}

// row softmax over R valid cols of an ld-wide row; zero the pad tail.
__global__ __launch_bounds__(256)
void softmax_rows(float* __restrict__ data, int R, int ld, int rowsPerBatch,
                  long long batchStride) {
  const int gr = blockIdx.x;
  float* p = data + (size_t)(gr / rowsPerBatch) * batchStride +
             (size_t)(gr % rowsPerBatch) * ld;
  __shared__ float sm[8];
  const int tid = threadIdx.x, wid = tid >> 5, lane = tid & 31;
  float mx = -1e30f;
  for (int i = tid; i < R; i += blockDim.x) mx = fmaxf(mx, p[i]);
  for (int m = 16; m >= 1; m >>= 1) mx = fmaxf(mx, __shfl_xor(mx, m));
  if (lane == 0) sm[wid] = mx;
  __syncthreads();
  float bm = -1e30f;
  for (int i = 0; i < 8; i++) bm = fmaxf(bm, sm[i]);
  __syncthreads();
  float sum = 0.f;
  for (int i = tid; i < R; i += blockDim.x) {
    float e = __expf(p[i] - bm);
    p[i] = e; sum += e;
  }
  for (int m = 16; m >= 1; m >>= 1) sum += __shfl_xor(sum, m);
  if (lane == 0) sm[wid] = sum;
  __syncthreads();
  float bs = 0.f;
  for (int i = 0; i < 8; i++) bs += sm[i];
  float inv = 1.0f / bs;
  for (int i = tid; i < R; i += blockDim.x) p[i] *= inv;
  for (int i = R + tid; i < ld; i += blockDim.x) p[i] = 0.0f;   // zero K-pad
}

// z = q + alpha * z (in place)
__global__ void add_scaled(const float* __restrict__ q, const float* __restrict__ alpha,
                           float* __restrict__ z, size_t n) {
  size_t i = (size_t)blockIdx.x * blockDim.x + threadIdx.x;
  if (i < n) z[i] = q[i] + alpha[0] * z[i];
}

// SRU recurrence: thread per (b, h), sequential over n
__global__ void sru_scan(const float* __restrict__ U, const float* __restrict__ xt,
                         const float* __restrict__ vf, const float* __restrict__ vr,
                         const float* __restrict__ bf, const float* __restrict__ br,
                         float* __restrict__ Hs) {
  int idx = blockIdx.x * blockDim.x + threadIdx.x;
  if (idx >= B_SZ * H_DIM) return;
  int b = idx / H_DIM, h = idx % H_DIM;
  float c = 0.f;
  const float Vf = vf[h], Vr = vr[h], Bf = bf[h], Br = br[h];
  for (int n = 0; n < NSEG; n++) {
    size_t base = (size_t)(b * NSEG + n);
    const float* u = U + base * U3;
    float u0 = u[h], u1 = u[h + H_DIM], u2 = u[h + 2 * H_DIM];
    float xv = xt[base * D_DIM + h];
    float f = 1.0f / (1.0f + __expf(-(u0 + Vf * c + Bf)));
    float cn = f * c + (1.0f - f) * u2;
    float r = 1.0f / (1.0f + __expf(-(u1 + Vr * c + Br)));
    Hs[base * H_DIM + h] = r * cn + (1.0f - r) * xv;
    c = cn;
  }
}

// y = LayerNorm(a + r) * g + beta ; one wave32 per row of D_DIM=512
__global__ __launch_bounds__(256)
void residual_ln(const float* __restrict__ a, const float* __restrict__ r,
                 const float* __restrict__ g, const float* __restrict__ beta,
                 float* __restrict__ y, int rows) {
  const int wpb = blockDim.x >> 5;
  const int row = blockIdx.x * wpb + (threadIdx.x >> 5);
  if (row >= rows) return;
  const int lane = threadIdx.x & 31;
  const float* pa = a + (size_t)row * D_DIM;
  const float* pr = r + (size_t)row * D_DIM;
  float v[16];
  float s = 0.f;
#pragma unroll
  for (int i = 0; i < 16; i++) {
    int d = lane + 32 * i;
    v[i] = pa[d] + pr[d];
    s += v[i];
  }
  for (int m = 16; m >= 1; m >>= 1) s += __shfl_xor(s, m);
  float mean = s * (1.0f / D_DIM);
  float s2 = 0.f;
#pragma unroll
  for (int i = 0; i < 16; i++) { float t = v[i] - mean; s2 += t * t; }
  for (int m = 16; m >= 1; m >>= 1) s2 += __shfl_xor(s2, m);
  float rs = rsqrtf(s2 * (1.0f / D_DIM) + EPS_F);
  float* py = y + (size_t)row * D_DIM;
#pragma unroll
  for (int i = 0; i < 16; i++) {
    int d = lane + 32 * i;
    py[d] = (v[i] - mean) * rs * g[d] + beta[d];
  }
}

// out[b,d] = mean_n in[b,n,d]
__global__ void mean_rows(const float* __restrict__ in, float* __restrict__ out, int R) {
  int idx = blockIdx.x * blockDim.x + threadIdx.x;
  if (idx >= B_SZ * D_DIM) return;
  int b = idx / D_DIM, d = idx % D_DIM;
  const float* p = in + (size_t)b * R * D_DIM + d;
  float s = 0.f;
  for (int n = 0; n < R; n++) s += p[(size_t)n * D_DIM];
  out[idx] = s / (float)R;
}

// cosmat[k][n] = cos(2*pi*k*n/511) for n<511, 0 in pad col (ld = 512)
__global__ void cosmat_init(float* __restrict__ cm) {
  int idx = blockIdx.x * blockDim.x + threadIdx.x;
  if (idx >= NF * NSEGP) return;
  int k = idx / NSEGP, n = idx % NSEGP;
  float v = 0.0f;
  if (n < NSEG) {
    long long ph = ((long long)k * n) % NSEG;
    v = cosf(6.283185307179586f * (float)ph / (float)NSEG);
  }
  cm[idx] = v;
}

// ---------------------------------------------------------------------------
// Host side
// ---------------------------------------------------------------------------
enum {
  IN_X = 0, IN_REV_W, IN_REV_B,
  IN_TB_WQ, IN_TB_WK, IN_TB_WV, IN_TB_WU, IN_TB_BU, IN_TB_ALPHA,
  IN_TB_VF, IN_TB_VR, IN_TB_BF, IN_TB_BR, IN_TB_WOUT,
  IN_TB_G1, IN_TB_B1N, IN_TB_G2, IN_TB_B2N,
  IN_TB_W1, IN_TB_B1, IN_TB_W2, IN_TB_B2,
  IN_FB_WQ, IN_FB_WK, IN_FB_WV, IN_FB_A, IN_FB_WO, IN_FB_BO,
  IN_FB_G1, IN_FB_B1N, IN_FB_G2, IN_FB_B2N,
  IN_FB_W1, IN_FB_B1, IN_FB_W2, IN_FB_B2,
  IN_TA_W, IN_TA_B, IN_FA_W, IN_FA_B, IN_DEC_W, IN_DEC_B
};

static float* wsAllocF(void* ws, size_t& off, size_t nfloats) {
  float* p = (float*)((char*)ws + off);
  off += nfloats * sizeof(float);
  off = (off + 255) & ~(size_t)255;
  return p;
}
static unsigned short* wsAllocH(void* ws, size_t& off, size_t nelems) {
  unsigned short* p = (unsigned short*)((char*)ws + off);
  off += nelems * sizeof(unsigned short);
  off = (off + 255) & ~(size_t)255;
  return p;
}

extern "C" void kernel_launch(void* const* d_in, const int* in_sizes, int n_in,
                              void* d_out, int out_size, void* d_ws, size_t ws_size,
                              hipStream_t stream) {
  (void)in_sizes; (void)n_in; (void)out_size; (void)ws_size;
  const float* x = (const float*)d_in[IN_X];
  auto F = [&](int i) { return (const float*)d_in[i]; };

  const size_t SZ_DATA = (size_t)B_SZ * NSEG * D_DIM;     // 8,372,224 live elems
  const size_t SZ_BUF  = (size_t)B_SZ * NSEGP * D_DIM;    // 8,388,608 (pad slack)
  const size_t SZ_BIG  = (size_t)B_SZ * NSEG * FF;        // 33,488,896

  size_t off = 0;
  float* tok  = wsAllocF(d_ws, off, SZ_BUF);
  float* bA   = wsAllocF(d_ws, off, SZ_BUF);
  float* bB   = wsAllocF(d_ws, off, SZ_BUF);
  float* bC   = wsAllocF(d_ws, off, SZ_BUF);
  float* bD   = wsAllocF(d_ws, off, SZ_BUF);  // attn probs: ld 512, slab 512x512
  float* bE   = wsAllocF(d_ws, off, SZ_BUF);
  float* big  = wsAllocF(d_ws, off, SZ_BIG);  // U / ffn-h / scores / hf
  float* cosm = wsAllocF(d_ws, off, (size_t)NF * NSEGP);
  float* scA  = wsAllocF(d_ws, off, B_SZ * C_SZ);
  float* shA  = wsAllocF(d_ws, off, B_SZ * C_SZ);
  float* tmean= wsAllocF(d_ws, off, B_SZ * D_DIM);
  float* fmean= wsAllocF(d_ws, off, B_SZ * D_DIM);
  float* feats= wsAllocF(d_ws, off, (size_t)B_SZ * 2 * D_DIM);

  // bf16 weight shadows (for async B staging)
  const int SQ = H_DIM * D_DIM;            // 262144
  unsigned short* wq16  = wsAllocH(d_ws, off, SQ);
  unsigned short* wk16  = wsAllocH(d_ws, off, SQ);
  unsigned short* wv16  = wsAllocH(d_ws, off, SQ);
  unsigned short* wu16  = wsAllocH(d_ws, off, (size_t)U3 * H_DIM);
  unsigned short* w116  = wsAllocH(d_ws, off, (size_t)FF * D_DIM);
  unsigned short* w216  = wsAllocH(d_ws, off, (size_t)D_DIM * FF);
  unsigned short* fwq16 = wsAllocH(d_ws, off, SQ);
  unsigned short* fwk16 = wsAllocH(d_ws, off, SQ);
  unsigned short* fwv16 = wsAllocH(d_ws, off, SQ);
  unsigned short* fwo16 = wsAllocH(d_ws, off, SQ);
  unsigned short* fw116 = wsAllocH(d_ws, off, (size_t)FF * D_DIM);
  unsigned short* fw216 = wsAllocH(d_ws, off, (size_t)D_DIM * FF);
  unsigned short* ta16  = wsAllocH(d_ws, off, SQ);
  unsigned short* fa16  = wsAllocH(d_ws, off, SQ);

  const int MT = B_SZ * NSEG;   // 16352 flat token rows
  const int MF = B_SZ * NF;     // 8192 flat freq rows
  const float rsH = 0.04419417382415922f;  // 1/sqrt(512)

  auto grid = [](int M, int Nd, int bat) { return dim3((Nd + BN - 1) / BN, (M + BM - 1) / BM, bat); };
  const dim3 blk(256);
  auto cvt = [&](int idx, unsigned short* dst, int n) {
    cvt_bf16<<<(n + 255) / 256, 256, 0, stream>>>(F(idx), dst, n);
  };

  // ---- weight pre-conversion to bf16 (one pass over ~16 MB) ----
  cvt(IN_TB_WQ, wq16, SQ);           cvt(IN_TB_WK, wk16, SQ);
  cvt(IN_TB_WV, wv16, SQ);           cvt(IN_TB_WU, wu16, U3 * H_DIM);
  cvt(IN_TB_W1, w116, FF * D_DIM);   cvt(IN_TB_W2, w216, D_DIM * FF);
  cvt(IN_FB_WQ, fwq16, SQ);          cvt(IN_FB_WK, fwk16, SQ);
  cvt(IN_FB_WV, fwv16, SQ);          cvt(IN_FB_WO, fwo16, SQ);
  cvt(IN_FB_W1, fw116, FF * D_DIM);  cvt(IN_FB_W2, fw216, D_DIM * FF);
  cvt(IN_TA_W, ta16, SQ);            cvt(IN_FA_W, fa16, SQ);

  // ---- stage 0: RevIN + segmentation ----
  revin_stats<<<B_SZ * C_SZ, 256, 0, stream>>>(x, F(IN_REV_W), F(IN_REV_B), scA, shA);
  build_tokens<<<(int)((SZ_DATA + 255) / 256), 256, 0, stream>>>(x, scA, shA, tok);

  // ---- time branch: SRU mixer ----
  gemm_wmma_bf16<true, 0, true><<<grid(MT, H_DIM, 1), blk, 0, stream>>>(
      tok, wq16, nullptr, bA, MT, H_DIM, D_DIM, D_DIM, D_DIM, H_DIM, 0, 0, 0, 1.0f);             // Q
  gemm_wmma_bf16<true, 0, true><<<grid(MT, H_DIM, 1), blk, 0, stream>>>(
      bA, wk16, nullptr, bB, MT, H_DIM, H_DIM, H_DIM, H_DIM, H_DIM, 0, 0, 0, 1.0f);              // Km
  gemm_wmma_bf16<true, 0, true><<<grid(MT, H_DIM, 1), blk, 0, stream>>>(
      bA, wv16, nullptr, bC, MT, H_DIM, H_DIM, H_DIM, H_DIM, H_DIM, 0, 0, 0, 1.0f);              // Vm
  gemm_wmma_bf16<true, 0, false><<<grid(NSEG, NSEG, B_SZ), blk, 0, stream>>>(
      bA, bB, nullptr, bD, NSEG, NSEG, H_DIM, H_DIM, H_DIM, NSEGP,
      (long long)NSEG * H_DIM, (long long)NSEG * H_DIM, (long long)NSEGP * NSEGP, rsH);          // logits
  softmax_rows<<<B_SZ * NSEG, 256, 0, stream>>>(bD, NSEG, NSEGP, NSEG,
                                                (long long)NSEGP * NSEGP);
  gemm_wmma_bf16<false, 0, false><<<grid(NSEG, H_DIM, B_SZ), blk, 0, stream>>>(
      bD, bC, nullptr, bE, NSEG, H_DIM, NSEGP, NSEGP, H_DIM, H_DIM,
      (long long)NSEGP * NSEGP, (long long)NSEG * H_DIM, (long long)NSEG * H_DIM, 1.0f);         // Fm
  add_scaled<<<(int)((SZ_DATA + 255) / 256), 256, 0, stream>>>(bA, F(IN_TB_ALPHA), bE, SZ_DATA); // Z
  gemm_wmma_bf16<true, 1, true><<<grid(MT, U3, 1), blk, 0, stream>>>(
      bE, wu16, F(IN_TB_BU), big, MT, U3, H_DIM, H_DIM, H_DIM, U3, 0, 0, 0, 1.0f);               // U
  sru_scan<<<(B_SZ * H_DIM) / 256, 256, 0, stream>>>(
      big, tok, F(IN_TB_VF), F(IN_TB_VR), F(IN_TB_BF), F(IN_TB_BR), bA);                         // Hs
  gemm_wmma_bf16<false, 0, false><<<grid(MT, D_DIM, 1), blk, 0, stream>>>(
      bA, F(IN_TB_WOUT), nullptr, bB, MT, D_DIM, H_DIM, H_DIM, D_DIM, D_DIM, 0, 0, 0, 1.0f);     // t_mix
  residual_ln<<<(MT + 7) / 8, 256, 0, stream>>>(tok, bB, F(IN_TB_G1), F(IN_TB_B1N), bC, MT);     // x1
  gemm_wmma_bf16<true, 2, true><<<grid(MT, FF, 1), blk, 0, stream>>>(
      bC, w116, F(IN_TB_B1), big, MT, FF, D_DIM, D_DIM, D_DIM, FF, 0, 0, 0, 1.0f);               // gelu h
  gemm_wmma_bf16<true, 1, true><<<grid(MT, D_DIM, 1), blk, 0, stream>>>(
      big, w216, F(IN_TB_B2), bD, MT, D_DIM, FF, FF, FF, D_DIM, 0, 0, 0, 1.0f);                  // ffn out
  residual_ln<<<(MT + 7) / 8, 256, 0, stream>>>(bC, bD, F(IN_TB_G2), F(IN_TB_B2N), bE, MT);      // t_out
  mean_rows<<<(B_SZ * D_DIM) / 256, 256, 0, stream>>>(bE, tmean, NSEG);
  gemm_wmma_bf16<true, 1, true><<<grid(B_SZ, D_DIM, 1), blk, 0, stream>>>(
      tmean, ta16, F(IN_TA_B), feats, B_SZ, D_DIM, D_DIM, D_DIM, D_DIM, 2 * D_DIM,
      0, 0, 0, 1.0f);                                                                            // t_feat

  // ---- frequency branch: rfft (as cos-GEMM, K padded to 512) + linattn ----
  cosmat_init<<<(NF * NSEGP + 255) / 256, 256, 0, stream>>>(cosm);
  gemm_wmma_bf16<false, 0, false><<<grid(NF, D_DIM, B_SZ), blk, 0, stream>>>(
      cosm, tok, nullptr, bA, NF, D_DIM, NSEGP, NSEGP, D_DIM, D_DIM,
      0, (long long)NSEG * D_DIM, (long long)NF * D_DIM, 1.0f);                                  // ftok
  gemm_wmma_bf16<true, 0, true><<<grid(MF, D_DIM, 1), blk, 0, stream>>>(
      bA, fwq16, nullptr, bB, MF, D_DIM, D_DIM, D_DIM, D_DIM, D_DIM, 0, 0, 0, 1.0f);             // Q2
  gemm_wmma_bf16<true, 0, true><<<grid(MF, D_DIM, 1), blk, 0, stream>>>(
      bA, fwk16, nullptr, bC, MF, D_DIM, D_DIM, D_DIM, D_DIM, D_DIM, 0, 0, 0, 1.0f);             // Kx
  gemm_wmma_bf16<true, 0, true><<<grid(MF, D_DIM, 1), blk, 0, stream>>>(
      bA, fwv16, nullptr, bD, MF, D_DIM, D_DIM, D_DIM, D_DIM, D_DIM, 0, 0, 0, 1.0f);             // V
  gemm_wmma_bf16<false, 0, false><<<grid(MF, D_DIM, 1), blk, 0, stream>>>(
      bC, F(IN_FB_A), nullptr, bE, MF, D_DIM, D_DIM, D_DIM, D_DIM, D_DIM, 0, 0, 0, 1.0f);        // K_hat
  gemm_wmma_bf16<false, 0, false><<<grid(MF, D_DIM, 1), blk, 0, stream>>>(
      bD, F(IN_FB_A), nullptr, tok, MF, D_DIM, D_DIM, D_DIM, D_DIM, D_DIM, 0, 0, 0, 1.0f);       // V_hat
  gemm_wmma_bf16<true, 0, false><<<grid(NF, NF, B_SZ), blk, 0, stream>>>(
      bB, bE, nullptr, big, NF, NF, D_DIM, D_DIM, D_DIM, NF,
      (long long)NF * D_DIM, (long long)NF * D_DIM, (long long)NF * NF, rsH);                    // scores
  softmax_rows<<<B_SZ * NF, 256, 0, stream>>>(big, NF, NF, NF, (long long)NF * NF);
  gemm_wmma_bf16<false, 0, false><<<grid(NF, D_DIM, B_SZ), blk, 0, stream>>>(
      big, tok, nullptr, bC, NF, D_DIM, NF, NF, D_DIM, D_DIM,
      (long long)NF * NF, (long long)NF * D_DIM, (long long)NF * D_DIM, 1.0f);                   // SV
  gemm_wmma_bf16<true, 1, true><<<grid(MF, D_DIM, 1), blk, 0, stream>>>(
      bC, fwo16, F(IN_FB_BO), bD, MF, D_DIM, D_DIM, D_DIM, D_DIM, D_DIM, 0, 0, 0, 1.0f);         // f_mix
  residual_ln<<<(MF + 7) / 8, 256, 0, stream>>>(bA, bD, F(IN_FB_G1), F(IN_FB_B1N), bB, MF);      // xf1
  gemm_wmma_bf16<true, 2, true><<<grid(MF, FF, 1), blk, 0, stream>>>(
      bB, fw116, F(IN_FB_B1), big, MF, FF, D_DIM, D_DIM, D_DIM, FF, 0, 0, 0, 1.0f);              // gelu hf
  gemm_wmma_bf16<true, 1, true><<<grid(MF, D_DIM, 1), blk, 0, stream>>>(
      big, fw216, F(IN_FB_B2), bE, MF, D_DIM, FF, FF, FF, D_DIM, 0, 0, 0, 1.0f);                 // ffn out
  residual_ln<<<(MF + 7) / 8, 256, 0, stream>>>(bB, bE, F(IN_FB_G2), F(IN_FB_B2N), bC, MF);      // f_out
  mean_rows<<<(B_SZ * D_DIM) / 256, 256, 0, stream>>>(bC, fmean, NF);
  gemm_wmma_bf16<true, 1, true><<<grid(B_SZ, D_DIM, 1), blk, 0, stream>>>(
      fmean, fa16, F(IN_FA_B), feats + D_DIM, B_SZ, D_DIM, D_DIM, D_DIM, D_DIM, 2 * D_DIM,
      0, 0, 0, 1.0f);                                                                            // f_feat

  // ---- decode head: (32 x 1024) @ (32 x 1024)^T + b (f32 B path, N edge) ----
  gemm_wmma_bf16<true, 1, false><<<grid(B_SZ, OUTD, 1), blk, 0, stream>>>(
      feats, F(IN_DEC_W), F(IN_DEC_B), (float*)d_out, B_SZ, OUTD, 2 * D_DIM,
      2 * D_DIM, 2 * D_DIM, OUTD, 0, 0, 0, 1.0f);
}
